// SepConvModel_81088982548607
// MI455X (gfx1250) — compile-verified, matrix operand order checked by reading, not run on hardware
//
#include <hip/hip_runtime.h>
#include <math.h>

typedef __attribute__((ext_vector_type(16))) __bf16 v16bf;
typedef __attribute__((ext_vector_type(8)))  float  v8f;
typedef __attribute__((ext_vector_type(4)))  int    v4i;

typedef __attribute__((address_space(1))) v4i GblV4i;
typedef __attribute__((address_space(3))) v4i LdsV4i;

#define BB 2
#define HH 256
#define WW 448
#define MPIX (BB*HH*WW)   // 229376 = 128*1792 = 256*896
#define KTAP 51

// ---- CDNA5 async global->LDS copy (ASYNCcnt path), with guarded fallback
__device__ __forceinline__ void async_copy16(const void* g, void* l) {
#if __has_builtin(__builtin_amdgcn_global_load_async_to_lds_b128)
  __builtin_amdgcn_global_load_async_to_lds_b128((GblV4i*)g, (LdsV4i*)l, 0, 0);
#else
  *(uint4*)l = *(const uint4*)g;
#endif
}

__device__ __forceinline__ void wait_async0() {
#if __has_builtin(__builtin_amdgcn_s_wait_asynccnt)
  __builtin_amdgcn_s_wait_asynccnt(0);
#else
  asm volatile("s_wait_asynccnt 0x0" ::: "memory");
#endif
}

// ---------------- input packing: NCHW f32 frames -> NHWC bf16 [M,32] (6 real ch + zero pad)
__global__ void pack_input_k(const float* __restrict__ f1, const float* __restrict__ f2,
                             __bf16* __restrict__ xout) {
  int m = blockIdx.x * 256 + threadIdx.x;
  if (m >= MPIX) return;
  int b = m / (HH * WW), r = m % (HH * WW);
  size_t plane = (size_t)HH * WW;
  const float* p1 = f1 + (size_t)b * 3 * plane + r;
  const float* p2 = f2 + (size_t)b * 3 * plane + r;
  __bf16* o = xout + (size_t)m * 32;
  #pragma unroll
  for (int c = 0; c < 32; ++c) {
    float v = 0.f;
    if (c < 3) v = p1[(size_t)c * plane];
    else if (c < 6) v = p2[(size_t)(c - 3) * plane];
    o[c] = (__bf16)v;
  }
}

// ---------------- weight packing: OIHW f32 -> [tap][cin/32][CoutP][32] bf16 (zero padded)
// n-major with 32 contiguous K per row: B tiles become straight 16B global->LDS copies.
__global__ void pack_w_k(const float* __restrict__ w, __bf16* __restrict__ wp,
                         int CINR, int CINP, int COUT, int COUTP, int KK) {
  int idx = blockIdx.x * 256 + threadIdx.x;
  int total = KK * CINP * COUTP;
  if (idx >= total) return;
  int kc  = idx & 31;
  int co  = (idx >> 5) % COUTP;
  int rem = (idx >> 5) / COUTP;
  int cb  = rem % (CINP >> 5);
  int tap = rem / (CINP >> 5);
  int ci  = cb * 32 + kc;
  float v = 0.f;
  if (ci < CINR && co < COUT)
    v = w[((size_t)co * CINR + ci) * KK + tap];
  wp[idx] = (__bf16)v;
}

// ---------------- implicit-GEMM conv via v_wmma_f32_16x16x32_bf16
// act: bf16 NHWC [M, CIN]; wp: bf16 [KK][CIN/32][COUTP][32]; out: [M, COUTP] bf16 or f32
template<int CIN, int COUT, int COUTP, int KS, bool RELU, bool OUTF32>
__global__ __launch_bounds__(256)
void conv_gemm_k(const __bf16* __restrict__ act, const __bf16* __restrict__ wp,
                 const float* __restrict__ bias, void* __restrict__ outv) {
  constexpr int BM = 128;
  constexpr int BN = (COUTP >= 64) ? 64 : COUTP;   // 32 or 64
  constexpr int KK = KS * KS;
  constexpr int PADK = KS / 2;
  constexpr int WAVES_N = BN / 32;                 // 1 or 2
  constexpr int WAVES_M = 8 / WAVES_N;             // 8 or 4
  constexpr int TM = BM / (16 * WAVES_M);          // 1 or 2
  constexpr int TN = 2;
  constexpr int LDA = 40;                          // 32 + 8 pad -> conflict-free DS reads
  constexpr int LDB = 40;

  __shared__ __align__(16) __bf16 Alds[BM * LDA];
  __shared__ __align__(16) __bf16 Blds[BN * LDB];

  const int t    = threadIdx.x;
  const int lane = t & 31;
  const int wave = t >> 5;
  const int wn   = wave % WAVES_N;
  const int wm   = wave / WAVES_N;
  const int mBlock = blockIdx.x * BM;
  const int nBlock = blockIdx.y * BN;

  // A-tile loader mapping: each thread owns one (row, 16-col half) forever
  const int ar  = t >> 1;           // 0..127
  const int ac  = (t & 1) * 16;     // 0 or 16
  const int am  = mBlock + ar;
  const int ab  = am / (HH * WW);
  const int arem = am % (HH * WW);
  const int ay  = arem / WW;
  const int ax  = arem % WW;

  // B-tile loader mapping: row n_local (32 contiguous K), 16B quarter
  const int bn  = t >> 2;
  const int bh  = (t & 3) * 8;
  const bool bact = (t < BN * 4);

  v8f zero = {};
  v8f acc[TM][TN];
  #pragma unroll
  for (int i = 0; i < TM; ++i)
    #pragma unroll
    for (int j = 0; j < TN; ++j)
      acc[i][j] = zero;

  const int lr = lane & 15;
  const int kb = (lane >> 4) * 8;   // K-group base per ISA 16-bit A/B layout

  #pragma unroll 1
  for (int tap = 0; tap < KK; ++tap) {
    const int py = ay + tap / KS - PADK;
    const int px = ax + tap % KS - PADK;
    const bool valid = (py >= 0) && (py < HH) && (px >= 0) && (px < WW);
    const __bf16* asrc = act + ((size_t)((ab * HH + py) * WW + px)) * CIN + ac;
    const __bf16* bsrc = wp + (((size_t)tap * (CIN / 32)) * COUTP + nBlock + bn) * 32 + bh;

    #pragma unroll 1
    for (int ci0 = 0; ci0 < CIN; ci0 += 32) {
      __syncthreads();   // previous iteration's fragment reads done before LDS overwrite
      // ---- A tile: async global->LDS for in-image rows, zero-fill otherwise
      if (valid) {
        async_copy16(asrc + ci0,     &Alds[ar * LDA + ac]);
        async_copy16(asrc + ci0 + 8, &Alds[ar * LDA + ac + 8]);
      } else {
        uint4 z = {0u, 0u, 0u, 0u};
        *(uint4*)&Alds[ar * LDA + ac]     = z;
        *(uint4*)&Alds[ar * LDA + ac + 8] = z;
      }
      // ---- B tile: async global->LDS (already n-major, K-contiguous in wp)
      if (bact)
        async_copy16(bsrc + ((size_t)(ci0 >> 5)) * COUTP * 32, &Blds[bn * LDB + bh]);
      wait_async0();
      __syncthreads();

      // ---- fragments (two 16B DS reads each, per ISA 16-bit 16x32 layout)
      v16bf afrag[TM], bfrag[TN];
      #pragma unroll
      for (int i = 0; i < TM; ++i) {
        const int row = wm * (TM * 16) + i * 16 + lr;
        union { uint4 q[2]; v16bf v; } u;
        u.q[0] = *(const uint4*)&Alds[row * LDA + kb];
        u.q[1] = *(const uint4*)&Alds[row * LDA + kb + 16];
        afrag[i] = u.v;
      }
      #pragma unroll
      for (int j = 0; j < TN; ++j) {
        const int row = wn * 32 + j * 16 + lr;
        union { uint4 q[2]; v16bf v; } u;
        u.q[0] = *(const uint4*)&Blds[row * LDB + kb];
        u.q[1] = *(const uint4*)&Blds[row * LDB + kb + 16];
        bfrag[j] = u.v;
      }
      #pragma unroll
      for (int i = 0; i < TM; ++i)
        #pragma unroll
        for (int j = 0; j < TN; ++j)
          acc[i][j] = __builtin_amdgcn_wmma_f32_16x16x32_bf16(
              false, afrag[i], false, bfrag[j], (short)0, acc[i][j], false, false);
    }
  }

  // ---- epilogue: bias + ReLU, store NHWC
  const int lh = lane >> 4;
  #pragma unroll
  for (int j = 0; j < TN; ++j) {
    const int n = nBlock + wn * 32 + j * 16 + lr;
    const float bv = (n < COUT) ? bias[n] : 0.f;
    #pragma unroll
    for (int i = 0; i < TM; ++i) {
      const int mrow0 = mBlock + wm * (TM * 16) + i * 16 + 8 * lh;
      #pragma unroll
      for (int r = 0; r < 8; ++r) {
        float v = acc[i][j][r] + bv;
        if (RELU) v = fmaxf(v, 0.f);
        const size_t off = (size_t)(mrow0 + r) * COUTP + n;
        if (OUTF32) ((float*)outv)[off] = v;
        else        ((__bf16*)outv)[off] = (__bf16)v;
      }
    }
  }
}

// ---------------- channel softmax over 51 of 64 padded logits -> NCHW f32 kernels
__global__ void softmax_k(const float* __restrict__ logits, float* __restrict__ out) {
  int m = blockIdx.x * 256 + threadIdx.x;
  if (m >= MPIX) return;
  int b = m / (HH * WW), r = m % (HH * WW);
  const float* lp = logits + (size_t)m * 64;
  float mx = lp[0];
  #pragma unroll 1
  for (int k = 1; k < KTAP; ++k) mx = fmaxf(mx, lp[k]);
  float s = 0.f;
  float e[KTAP];
  #pragma unroll 1
  for (int k = 0; k < KTAP; ++k) { e[k] = __expf(lp[k] - mx); s += e[k]; }
  float inv = 1.f / s;
  float* op = out + (size_t)b * KTAP * HH * WW + r;
  #pragma unroll 1
  for (int k = 0; k < KTAP; ++k) op[(size_t)k * HH * WW] = e[k] * inv;
}

// ---------------- separable filtering: vertical pass (zero-padded taps)
__global__ void sepconv_v_k(const float* __restrict__ frame, const float* __restrict__ vk,
                            float* __restrict__ out) {
  int idx = blockIdx.x * 256 + threadIdx.x;
  if (idx >= BB * 3 * HH * WW) return;
  int x = idx % WW; int t1 = idx / WW;
  int y = t1 % HH;  int t2 = t1 / HH;
  int c = t2 % 3;   int b  = t2 / 3;
  const float* kp = vk + ((size_t)b * KTAP * HH + y) * WW + x;
  const float* fp = frame + ((size_t)(b * 3 + c) * HH) * WW + x;
  float acc = 0.f;
  #pragma unroll 1
  for (int k = 0; k < KTAP; ++k) {
    int yy = y + k - KTAP / 2;
    if (yy >= 0 && yy < HH)
      acc += fp[(size_t)yy * WW] * kp[(size_t)k * HH * WW];
  }
  out[idx] = acc;
}

// ---------------- horizontal pass + 0.5/0.5 blend
__global__ void sepconv_h_k(const float* __restrict__ v1, const float* __restrict__ v2,
                            const float* __restrict__ hk, float* __restrict__ o_interp,
                            float* __restrict__ o_c1, float* __restrict__ o_c2) {
  int idx = blockIdx.x * 256 + threadIdx.x;
  if (idx >= BB * 3 * HH * WW) return;
  int x = idx % WW; int t1 = idx / WW;
  int y = t1 % HH;  int t2 = t1 / HH;
  int c = t2 % 3;   int b  = t2 / 3;
  const float* kp = hk + ((size_t)b * KTAP * HH + y) * WW + x;
  const float* p1 = v1 + ((size_t)(b * 3 + c) * HH + y) * WW;
  const float* p2 = v2 + ((size_t)(b * 3 + c) * HH + y) * WW;
  float a1 = 0.f, a2 = 0.f;
  #pragma unroll 1
  for (int k = 0; k < KTAP; ++k) {
    int xx = x + k - KTAP / 2;
    if (xx >= 0 && xx < WW) {
      float w = kp[(size_t)k * HH * WW];
      a1 += p1[xx] * w;
      a2 += p2[xx] * w;
    }
  }
  o_c1[idx] = a1;
  o_c2[idx] = a2;
  o_interp[idx] = 0.5f * (a1 + a2);
}

extern "C" void kernel_launch(void* const* d_in, const int* in_sizes, int n_in,
                              void* d_out, int out_size, void* d_ws, size_t ws_size,
                              hipStream_t stream) {
  const float* frame1 = (const float*)d_in[0];
  const float* frame2 = (const float*)d_in[1];
  const float* few[4] = {(const float*)d_in[2], (const float*)d_in[3],
                         (const float*)d_in[4], (const float*)d_in[5]};
  const float* feb[4] = {(const float*)d_in[6], (const float*)d_in[7],
                         (const float*)d_in[8], (const float*)d_in[9]};
  const float* vkw[3] = {(const float*)d_in[10], (const float*)d_in[11], (const float*)d_in[12]};
  const float* vkb[3] = {(const float*)d_in[13], (const float*)d_in[14], (const float*)d_in[15]};
  const float* hkw[3] = {(const float*)d_in[16], (const float*)d_in[17], (const float*)d_in[18]};
  const float* hkb[3] = {(const float*)d_in[19], (const float*)d_in[20], (const float*)d_in[21]};

  char* ws = (char*)d_ws;
  size_t off = 0;
  auto alloc = [&](size_t bytes) { size_t o = off; off = (off + bytes + 255) & ~(size_t)255; return o; };
  __bf16* X32 = (__bf16*)(ws + alloc((size_t)MPIX * 32 * 2));
  __bf16* B1  = (__bf16*)(ws + alloc((size_t)MPIX * 256 * 2));
  __bf16* B2  = (__bf16*)(ws + alloc((size_t)MPIX * 256 * 2));
  __bf16* FE  = (__bf16*)(ws + alloc((size_t)MPIX * 256 * 2));
  float*  LG  = (float*)(ws + alloc((size_t)MPIX * 64 * 4));
  __bf16* WP  = (__bf16*)(ws + alloc((size_t)49 * 128 * 256 * 2));
  float*  V1  = (float*)(ws + alloc((size_t)BB * 3 * HH * WW * 4));
  float*  V2  = (float*)(ws + alloc((size_t)BB * 3 * HH * WW * 4));

  float* out_interp = (float*)d_out;
  float* out_vk = out_interp + (size_t)BB * 3 * HH * WW;
  float* out_hk = out_vk + (size_t)BB * KTAP * HH * WW;
  float* out_c1 = out_hk + (size_t)BB * KTAP * HH * WW;
  float* out_c2 = out_c1 + (size_t)BB * 3 * HH * WW;

  dim3 blk(256);
  pack_input_k<<<MPIX / 256, blk, 0, stream>>>(frame1, frame2, X32);

  auto packw = [&](const float* w, int cinr, int cinp, int cout, int coutp, int kk) {
    int total = kk * cinp * coutp;
    pack_w_k<<<(total + 255) / 256, blk, 0, stream>>>(w, WP, cinr, cinp, cout, coutp, kk);
  };

  // feature extractor: 7x7 convs, pad 3, ReLU
  packw(few[0], 6, 32, 32, 32, 49);
  conv_gemm_k<32, 32, 32, 7, true, false><<<dim3(MPIX / 128, 1), blk, 0, stream>>>(X32, WP, feb[0], B1);
  packw(few[1], 32, 32, 64, 64, 49);
  conv_gemm_k<32, 64, 64, 7, true, false><<<dim3(MPIX / 128, 1), blk, 0, stream>>>(B1, WP, feb[1], B2);
  packw(few[2], 64, 64, 128, 128, 49);
  conv_gemm_k<64, 128, 128, 7, true, false><<<dim3(MPIX / 128, 2), blk, 0, stream>>>(B2, WP, feb[2], B1);
  packw(few[3], 128, 128, 256, 256, 49);
  conv_gemm_k<128, 256, 256, 7, true, false><<<dim3(MPIX / 128, 4), blk, 0, stream>>>(B1, WP, feb[3], FE);

  // kernel-prediction branches: 3x3 convs, pad 1
  for (int br = 0; br < 2; ++br) {
    const float* const* wlist = br ? hkw : vkw;
    const float* const* blist = br ? hkb : vkb;
    float* out_kern = br ? out_hk : out_vk;
    packw(wlist[0], 256, 256, 256, 256, 9);
    conv_gemm_k<256, 256, 256, 3, true, false><<<dim3(MPIX / 128, 4), blk, 0, stream>>>(FE, WP, blist[0], B1);
    packw(wlist[1], 256, 256, 256, 256, 9);
    conv_gemm_k<256, 256, 256, 3, true, false><<<dim3(MPIX / 128, 4), blk, 0, stream>>>(B1, WP, blist[1], B2);
    packw(wlist[2], 256, 256, 51, 64, 9);
    conv_gemm_k<256, 51, 64, 3, false, true><<<dim3(MPIX / 128, 1), blk, 0, stream>>>(B2, WP, blist[2], LG);
    softmax_k<<<MPIX / 256, blk, 0, stream>>>(LG, out_kern);
  }

  // separable filtering of both frames + blend
  int npix = BB * 3 * HH * WW;
  sepconv_v_k<<<npix / 256, blk, 0, stream>>>(frame1, out_vk, V1);
  sepconv_v_k<<<npix / 256, blk, 0, stream>>>(frame2, out_vk, V2);
  sepconv_h_k<<<npix / 256, blk, 0, stream>>>(V1, V2, out_hk, out_interp, out_c1, out_c2);
}
